// SiameseNetwork_10093173146117
// MI455X (gfx1250) — matrix operand order repaired and checked by reading, not compiled, not run on hardware
//
#include <hip/hip_runtime.h>

typedef __attribute__((ext_vector_type(16))) _Float16 v16h;
typedef __attribute__((ext_vector_type(8)))  _Float16 v8h;
typedef __attribute__((ext_vector_type(8)))  float    v8f;
typedef unsigned int u32x4 __attribute__((ext_vector_type(4)));
typedef int          i32x4 __attribute__((ext_vector_type(4)));
typedef int          i32x8 __attribute__((ext_vector_type(8)));
typedef __attribute__((address_space(3))) void lds_void_t;

namespace {

constexpr int H = 256, W = 256, HWp = H * W;   // 65536 pixels/image
constexpr int B = 8, R = 4, NIMG = B + B * R;  // 40 images through the CNN
constexpr float BN_EPS = 1e-5f;

// --------------------------------------------------------------------------
// prep: fold BN into (scale, shift); pack WMMA weights to f16 with the
// K-dimension reordered as k = tap*CIN + c (channel fastest) so A operands
// become contiguous b128 LDS reads.
// SC layout (floats): [0:16) sc1 [16:32) sh1 [32:64) sc2 [64:96) sh2
//                     [96:128) sc3 [128:160) sh3 [160] sc4 [161] sh4
// --------------------------------------------------------------------------
__global__ __launch_bounds__(256) void prep_kernel(
    const float* b1, const float* g1, const float* be1, const float* m1, const float* v1,
    const float* w2, const float* b2, const float* g2, const float* be2, const float* m2, const float* v2,
    const float* w3, const float* b3, const float* g3, const float* be3, const float* m3, const float* v3,
    const float* b4, const float* g4, const float* be4, const float* m4, const float* v4,
    _Float16* wpk2, _Float16* wpk3, float* SC)
{
  int t = threadIdx.x;
  if (t < 16) {
    float s = g1[t] * rsqrtf(v1[t] + BN_EPS);
    SC[t] = s;
    SC[16 + t] = be1[t] - m1[t] * s + b1[t] * s;
  }
  if (t < 32) {
    float s2 = g2[t] * rsqrtf(v2[t] + BN_EPS);
    SC[32 + t] = s2;
    SC[64 + t] = be2[t] - m2[t] * s2 + b2[t] * s2;
    float s3 = g3[t] * rsqrtf(v3[t] + BN_EPS);
    SC[96 + t] = s3;
    SC[128 + t] = be3[t] - m3[t] * s3 + b3[t] * s3;
  }
  if (t == 0) {
    float s4 = g4[0] * rsqrtf(v4[0] + BN_EPS);
    SC[160] = s4;
    SC[161] = be4[0] - m4[0] * s4 + b4[0] * s4;
  }
  // conv2: w2 [32][16][3][3] -> wpk2[n][tap*16 + c], K padded 144->160
  for (int e = t; e < 32 * 160; e += 256) {
    int n = e / 160, k = e - n * 160;
    int tap = k >> 4, c = k & 15;
    wpk2[e] = (tap < 9) ? (_Float16)w2[n * 144 + c * 9 + tap] : (_Float16)0.f;
  }
  // conv3: w3 [32][32][5][5] -> wpk3[n][tap*32 + c], K = 800 exactly
  for (int e = t; e < 32 * 800; e += 256) {
    int n = e / 800, k = e - n * 800;
    int tap = k >> 5, c = k & 31;
    wpk3[e] = (_Float16)w3[n * 800 + c * 25 + tap];
  }
}

// --------------------------------------------------------------------------
// stage: images 0..7 = conv0(gt) (3->1, 3x3, pad1, +b0); images 8..39 = refs.
// --------------------------------------------------------------------------
__global__ __launch_bounds__(256) void stage_kernel(
    const float* __restrict__ gt, const float* __restrict__ refs,
    const float* __restrict__ w0, const float* __restrict__ b0,
    float* __restrict__ act0)
{
  int idx = blockIdx.x * 256 + threadIdx.x;   // NIMG*HWp
  int img = idx >> 16, p = idx & (HWp - 1);
  if (img < B) {
    int h = p >> 8, w = p & 255;
    float acc = b0[0];
#pragma unroll
    for (int c = 0; c < 3; ++c)
#pragma unroll
      for (int r = 0; r < 3; ++r)
#pragma unroll
        for (int s = 0; s < 3; ++s) {
          int hh = h + r - 1, ww = w + s - 1;
          if (hh >= 0 && hh < H && ww >= 0 && ww < W)
            acc += gt[((img * 3 + c) << 16) + (hh << 8) + ww] * w0[(c * 3 + r) * 3 + s];
        }
    act0[idx] = acc;
  } else {
    act0[idx] = refs[idx - B * HWp];   // refs reshape(B*R,1,H,W) is contiguous
  }
}

// --------------------------------------------------------------------------
// conv1: 1->16, 3x3, pad1, BN + LeakyReLU, f32 in -> f16 out. Direct VALU.
// --------------------------------------------------------------------------
__global__ __launch_bounds__(256) void conv1_kernel(
    const float* __restrict__ act0, const float* __restrict__ w1,
    const float* __restrict__ SC, _Float16* __restrict__ act1)
{
  int idx = blockIdx.x * 256 + threadIdx.x;
  int img = idx >> 16, p = idx & (HWp - 1);
  int h = p >> 8, w = p & 255;
  float patch[9];
#pragma unroll
  for (int r = 0; r < 3; ++r)
#pragma unroll
    for (int s = 0; s < 3; ++s) {
      int hh = h + r - 1, ww = w + s - 1;
      patch[r * 3 + s] = (hh >= 0 && hh < H && ww >= 0 && ww < W)
                             ? act0[(img << 16) + (hh << 8) + ww] : 0.f;
    }
#pragma unroll
  for (int ch = 0; ch < 16; ++ch) {
    float acc = 0.f;
#pragma unroll
    for (int j = 0; j < 9; ++j) acc += patch[j] * w1[ch * 9 + j];
    float y = acc * SC[ch] + SC[16 + ch];
    y = (y >= 0.f) ? y : 0.01f * y;
    act1[((img * 16 + ch) << 16) + p] = (_Float16)y;
  }
}

// --------------------------------------------------------------------------
// Implicit-GEMM conv via V_WMMA_F32_16X16X32_F16.
// Block: 8 waves, one 128-pixel strip (within one image row) x 32 channels.
//  - Weights (32 x KPAD f16) -> LDS once via Tensor Data Mover (wave 0),
//    overlapped with input staging; completed with s_wait_tensorcnt.
//  - Input tile (CIN x KS rows x 132 cols, halo zero-padded) -> LDS once,
//    channel-interleaved so A operands are contiguous ds_load_b128.
//  - K order: k = tap*CIN + c; per 32-wide chunk: 2 A b128 + 4 B b128 + 2 WMMA.
// --------------------------------------------------------------------------
template <int CIN, int KS, int PAD, int KPAD>
__global__ __launch_bounds__(256) void conv_wmma_kernel(
    const _Float16* __restrict__ in,   // [NIMG][CIN][HWp]
    const _Float16* __restrict__ wpk,  // [32][KPAD], k = tap*CIN + c
    const float* __restrict__ sc, const float* __restrict__ sh,
    _Float16* __restrict__ out)        // [NIMG][32][HWp]
{
  constexpr int NTAP = KS * KS;
  constexpr int TPC = 32 / CIN;        // taps per 32-wide K chunk (1 or 2)
  constexpr int NCHUNK = KPAD / 32;
  constexpr int TW = 132;              // tile cols: ww = wbase-2+col

  __shared__ __align__(16) _Float16 Bs[32 * KPAD];
  __shared__ __align__(16) _Float16 Tile[KS * TW * CIN];

  const int tile  = blockIdx.x;            // NIMG * 512
  const int img   = tile >> 9;
  const int pbase = (tile & 511) << 7;
  const int h     = pbase >> 8;
  const int wbase = pbase & 255;           // 0 or 128
  const int tid   = threadIdx.x;
  const int wave  = tid >> 5, lane = tid & 31;
  const _Float16* inimg = in + (size_t)img * CIN * HWp;

  // ---- weights -> LDS ----
#if __has_builtin(__builtin_amdgcn_tensor_load_to_lds) && __has_builtin(__builtin_amdgcn_s_wait_tensorcnt)
  const bool tdm_wave = (wave == 0);
  if (tdm_wave) {
    unsigned long long gva = (unsigned long long)(const void*)wpk;
    unsigned lds_off = (unsigned)(unsigned long long)(lds_void_t*)&Bs[0];
    constexpr unsigned NEL = 32u * (unsigned)KPAD;   // total f16 elements
    u32x4 g0;
    g0.x = 1u;                                       // count=1 (valid, user mode)
    g0.y = lds_off;                                  // lds_addr (bytes)
    g0.z = (unsigned)gva;                            // global_addr[31:0]
    g0.w = (unsigned)((gva >> 32) & 0x01FFFFFFull) | 0x80000000u;  // [56:32] | type=2
    i32x8 g1;
    g1[0] = (int)(1u << 16);                         // data_size=1 (2 bytes)
    g1[1] = (int)((NEL & 0xFFFFu) << 16);            // tensor_dim0 lo16 -> [31:16]
    g1[2] = (int)((NEL >> 16) | (1u << 16));         // tensor_dim0 hi16 | tensor_dim1=1 lo
    g1[3] = (int)(NEL << 16);                        // tensor_dim1 hi=0 | tile_dim0=NEL
    g1[4] = (int)1u;                                 // tile_dim1=1, tile_dim2=0
    g1[5] = (int)NEL;                                // tensor_dim0_stride lo32
    g1[6] = 0;                                       // stride0 hi | stride1 lo
    g1[7] = 0;                                       // stride1 hi
    i32x4 g2 = {}; i32x4 g3 = {};
#if __clang_major__ >= 23
    i32x8 g4 = {};
    __builtin_amdgcn_tensor_load_to_lds(g0, g1, g2, g3, g4, 0);
#else
    __builtin_amdgcn_tensor_load_to_lds(g0, g1, g2, g3, 0);
#endif
  }
#else
  for (int e = tid; e < 32 * KPAD; e += 256) Bs[e] = wpk[e];
#endif

  // ---- input tile -> LDS, channel-interleaved, halo zero-padded ----
  // Tile[(r*TW + col)*CIN + c] holds input (c, h+r-PAD, wbase-2+col).
  constexpr int NPAIR = 66 * KS * CIN;   // 132 cols as 66 aligned b32 pairs
  for (int j = tid; j < NPAIR; j += 256) {
    int row = j / 66;                    // c*KS + r
    int pr  = j - row * 66;
    int c   = row / KS;
    int r   = row - c * KS;
    int hh  = h + r - PAD;
    int ww0 = wbase - 2 + pr * 2;        // even; pad boundaries are even too
    bool ok = (hh >= 0) & (hh < H) & (ww0 >= 0) & (ww0 < W);
    int hcl = hh < 0 ? 0 : (hh > H - 1 ? H - 1 : hh);
    int wcl = ww0 < 0 ? 0 : (ww0 > W - 2 ? W - 2 : ww0);
    unsigned v = *(const unsigned*)(inimg + (c << 16) + (hcl << 8) + wcl);
    v = ok ? v : 0u;
    int col0 = pr * 2;
    Tile[(r * TW + col0) * CIN + c]       = __builtin_bit_cast(_Float16, (unsigned short)(v & 0xFFFFu));
    Tile[(r * TW + col0 + 1) * CIN + c]   = __builtin_bit_cast(_Float16, (unsigned short)(v >> 16));
  }
#if __has_builtin(__builtin_amdgcn_tensor_load_to_lds) && __has_builtin(__builtin_amdgcn_s_wait_tensorcnt)
  if (tdm_wave) __builtin_amdgcn_s_wait_tensorcnt(0);
#endif
  __syncthreads();

  // ---- WMMA main loop ----
  const int wl  = (wave << 4) + (lane & 15);   // pixel column within strip
  const int kba = (lane < 16) ? 0 : 8;         // A half-select (ISA 16-bit A layout)
  const int nb  = lane & 15;                   // B: lane = N mod 16
  const int kbb = (lane < 16) ? 0 : 8;         // B: K half-select

  v8f c0 = {}, c1 = {};
  for (int kc = 0; kc < NCHUNK; ++kc) {
    v16h Aop;
#pragma unroll
    for (int g = 0; g < 2; ++g) {
      // element group g covers kk = kba + 16g + (0..7); k = tap*CIN + c
      int tl, cofs;
      if (CIN == 32) { tl = 0; cofs = kba + 16 * g; }
      else           { tl = g; cofs = kba; }
      int tap = kc * TPC + tl;
      v8h part = {};
      if (tap < NTAP) {
        int r = tap / KS, s = tap - (tap / KS) * KS;
        int col = wl + s - PAD + 2;
        part = *(const v8h*)&Tile[(r * TW + col) * CIN + cofs];
      }
#pragma unroll
      for (int i = 0; i < 8; ++i) Aop[8 * g + i] = part[i];
    }
    const _Float16* bp0 = &Bs[nb * KPAD + kc * 32 + kbb];
    const _Float16* bp1 = &Bs[(nb + 16) * KPAD + kc * 32 + kbb];
    v8h b0lo = *(const v8h*)bp0, b0hi = *(const v8h*)(bp0 + 16);
    v8h b1lo = *(const v8h*)bp1, b1hi = *(const v8h*)(bp1 + 16);
    v16h B0, B1;
#pragma unroll
    for (int i = 0; i < 8; ++i) {
      B0[i] = b0lo[i]; B0[8 + i] = b0hi[i];
      B1[i] = b1lo[i]; B1[8 + i] = b1hi[i];
    }
    c0 = __builtin_amdgcn_wmma_f32_16x16x32_f16(false, Aop, false, B0, (short)0, c0, false, false);
    c1 = __builtin_amdgcn_wmma_f32_16x16x32_f16(false, Aop, false, B1, (short)0, c1, false, false);
  }

  // ---- epilogue: BN + LeakyReLU. C/D: lane = N mod 16, VGPR v -> M = v(+8). ----
  const int n    = lane & 15;
  const int mofs = (lane < 16) ? 0 : 8;
  float s0 = sc[n], h0 = sh[n], s1 = sc[n + 16], h1 = sh[n + 16];
#pragma unroll
  for (int v = 0; v < 8; ++v) {
    int p = pbase + (wave << 4) + mofs + v;
    float y0 = c0[v] * s0 + h0;  y0 = (y0 >= 0.f) ? y0 : 0.01f * y0;
    float y1 = c1[v] * s1 + h1;  y1 = (y1 >= 0.f) ? y1 : 0.01f * y1;
    out[((size_t)(img * 32 + n) << 16) + p]      = (_Float16)y0;
    out[((size_t)(img * 32 + n + 16) << 16) + p] = (_Float16)y1;
  }
}

// --------------------------------------------------------------------------
// conv4: 32->1, 3x3, pad1, BN, +residual(act0), LeakyReLU -> f32 features.
// --------------------------------------------------------------------------
__global__ __launch_bounds__(256) void conv4_kernel(
    const _Float16* __restrict__ act3, const float* __restrict__ act0,
    const float* __restrict__ w4, const float* __restrict__ SC,
    float* __restrict__ feat)
{
  int idx = blockIdx.x * 256 + threadIdx.x;
  int img = idx >> 16, p = idx & (HWp - 1);
  int h = p >> 8, w = p & 255;
  float acc = 0.f;
  for (int c = 0; c < 32; ++c) {
    const _Float16* base = act3 + ((size_t)(img * 32 + c) << 16);
#pragma unroll
    for (int r = 0; r < 3; ++r)
#pragma unroll
      for (int s = 0; s < 3; ++s) {
        int hh = h + r - 1, ww = w + s - 1;
        if (hh >= 0 && hh < H && ww >= 0 && ww < W)
          acc += (float)base[(hh << 8) + ww] * w4[c * 9 + r * 3 + s];
      }
  }
  float y = acc * SC[160] + SC[161] + act0[idx];
  feat[idx] = (y >= 0.f) ? y : 0.01f * y;
}

// --------------------------------------------------------------------------
// finalize: |diff|, min/max over R, clip/threshold, first-max argmax, gather.
// d_out = dissim_map [8,4,256,256] ++ collage [8,1,256,256]  (f32)
// --------------------------------------------------------------------------
__global__ __launch_bounds__(256) void final_kernel(
    const float* __restrict__ feat, float* __restrict__ outp)
{
  int idx = blockIdx.x * 256 + threadIdx.x;   // B*HWp
  int b = idx >> 16, p = idx & (HWp - 1);
  float g = feat[(b << 16) + p];
  float rf[4], tot[4];
  float mn = 1e30f, mx = -1e30f;
#pragma unroll
  for (int r = 0; r < 4; ++r) {
    rf[r] = feat[((B + b * 4 + r) << 16) + p];
    tot[r] = fabsf(g - rf[r]);
    mn = fminf(mn, tot[r]);
    mx = fmaxf(mx, tot[r]);
  }
  float inv = 1.f / (mx - mn + 1e-8f);
  float best = -1.f; int bi = 0;
#pragma unroll
  for (int r = 0; r < 4; ++r) {
    float nrm = (tot[r] - mn) * inv;
    nrm = fminf(fmaxf(nrm, 0.1f), 1.0f);
    float d = (nrm > 0.5f) ? nrm : 0.f;
    outp[((b * 4 + r) << 16) + p] = d;
    if (d > best) { best = d; bi = r; }   // strict > == first-max (argmax)
  }
  outp[(size_t)B * R * HWp + (b << 16) + p] = rf[bi];
}

} // namespace

extern "C" void kernel_launch(void* const* d_in, const int* in_sizes, int n_in,
                              void* d_out, int out_size, void* d_ws, size_t ws_size,
                              hipStream_t stream) {
  (void)in_sizes; (void)n_in; (void)out_size; (void)ws_size;
  const float* gt  = (const float*)d_in[0];
  const float* refs= (const float*)d_in[1];
  const float* w0  = (const float*)d_in[2];
  const float* b0  = (const float*)d_in[3];
  const float* w1  = (const float*)d_in[4];
  const float* b1  = (const float*)d_in[5];
  const float* g1  = (const float*)d_in[6];
  const float* be1 = (const float*)d_in[7];
  const float* m1  = (const float*)d_in[8];
  const float* v1  = (const float*)d_in[9];
  const float* w2  = (const float*)d_in[10];
  const float* b2  = (const float*)d_in[11];
  const float* g2  = (const float*)d_in[12];
  const float* be2 = (const float*)d_in[13];
  const float* m2  = (const float*)d_in[14];
  const float* v2  = (const float*)d_in[15];
  const float* w3  = (const float*)d_in[16];
  const float* b3  = (const float*)d_in[17];
  const float* g3  = (const float*)d_in[18];
  const float* be3 = (const float*)d_in[19];
  const float* m3  = (const float*)d_in[20];
  const float* v3  = (const float*)d_in[21];
  const float* w4  = (const float*)d_in[22];
  const float* b4  = (const float*)d_in[23];
  const float* g4  = (const float*)d_in[24];
  const float* be4 = (const float*)d_in[25];
  const float* m4  = (const float*)d_in[26];
  const float* v4  = (const float*)d_in[27];

  char* ws = (char*)d_ws;
  size_t off = 0;
  auto carve = [&](size_t bytes) -> void* {
    void* p = ws + off;
    off = (off + bytes + 255) & ~(size_t)255;
    return p;
  };
  float*    act0 = (float*)   carve((size_t)NIMG * HWp * 4);        // 10.5 MB
  float*    feat = (float*)   carve((size_t)NIMG * HWp * 4);        // 10.5 MB
  _Float16* act1 = (_Float16*)carve((size_t)NIMG * 16 * HWp * 2);   // 84 MB
  _Float16* act2 = (_Float16*)carve((size_t)NIMG * 32 * HWp * 2);   // 168 MB
  _Float16* act3 = (_Float16*)carve((size_t)NIMG * 32 * HWp * 2);   // 168 MB
  _Float16* wpk2 = (_Float16*)carve(32 * 160 * 2);
  _Float16* wpk3 = (_Float16*)carve(32 * 800 * 2);
  float*    SC   = (float*)   carve(256 * 4);

  const int elemBlocks = NIMG * HWp / 256;

  prep_kernel<<<1, 256, 0, stream>>>(b1, g1, be1, m1, v1,
                                     w2, b2, g2, be2, m2, v2,
                                     w3, b3, g3, be3, m3, v3,
                                     b4, g4, be4, m4, v4,
                                     wpk2, wpk3, SC);
  stage_kernel<<<elemBlocks, 256, 0, stream>>>(gt, refs, w0, b0, act0);
  conv1_kernel<<<elemBlocks, 256, 0, stream>>>(act0, w1, SC, act1);
  conv_wmma_kernel<16, 3, 1, 160><<<NIMG * 512, 256, 0, stream>>>(act1, wpk2, SC + 32, SC + 64, act2);
  conv_wmma_kernel<32, 5, 2, 800><<<NIMG * 512, 256, 0, stream>>>(act2, wpk3, SC + 96, SC + 128, act3);
  conv4_kernel<<<elemBlocks, 256, 0, stream>>>(act3, act0, w4, SC, feat);
  final_kernel<<<B * HWp / 256, 256, 0, stream>>>(feat, (float*)d_out);
}
